// ImprovedTAPCell_52166672777401
// MI455X (gfx1250) — matrix operand order repaired and checked by reading, not compiled, not run on hardware
//
#include <hip/hip_runtime.h>
#include <hip/hip_bf16.h>

// ---------------------------------------------------------------------------
// Types
// ---------------------------------------------------------------------------
typedef unsigned short u16;
typedef __attribute__((ext_vector_type(8)))  unsigned short ushort8;
typedef __attribute__((ext_vector_type(16))) unsigned short ushort16;
typedef __attribute__((ext_vector_type(16))) __bf16        v16bf;
typedef __attribute__((ext_vector_type(8)))  float         v8f;

// f32 -> bf16 round-to-nearest-even
static __device__ __forceinline__ u16 f2bf(float f) {
    unsigned u = __builtin_bit_cast(unsigned, f);
    unsigned r = u + 0x7FFFu + ((u >> 16) & 1u);
    return (u16)(r >> 16);
}
static __device__ __forceinline__ float bf2f(u16 b) {
    return __builtin_bit_cast(float, ((unsigned)b) << 16);
}

// ---------------------------------------------------------------------------
// CDNA5 async global->LDS copy (ASYNCcnt-tracked), per ISA 15.18.3 op 98.
// Each lane moves 16 bytes from its global address to its LDS address.
// ---------------------------------------------------------------------------
static __device__ __forceinline__ void async_copy_b128(unsigned lds_addr, const u16* gptr) {
    asm volatile("global_load_async_to_lds_b128 %0, %1, off"
                 :: "v"(lds_addr), "v"(gptr)
                 : "memory");
}
static __device__ __forceinline__ void wait_async_le8() {
    asm volatile("s_wait_asynccnt 0x8" ::: "memory");
}
static __device__ __forceinline__ void wait_async_0() {
    asm volatile("s_wait_asynccnt 0x0" ::: "memory");
}

// ---------------------------------------------------------------------------
// WMMA fragment loaders (CDNA5 ISA 7.12.2 layouts, wave32)
// A 16x32 bf16: lane L (L<16) holds row M=L; elems 0..7 -> K = half*8 + 0..7,
//               elems 8..15 -> K = 16 + half*8 + 0..7   (half = lane>=16)
// B 32x16 bf16: lane L holds col N=L%16; elems 0..7 -> K = half*8 + 0..7,
//               elems 8..15 -> K = 16 + half*8 + 0..7
// Both assembled from two 16-byte loads (LDS rows are 16B-aligned).
// ---------------------------------------------------------------------------
static __device__ __forceinline__ v16bf load_a_frag(const u16* rowbase, int half) {
    ushort8 lo = *(const ushort8*)(rowbase + half * 8);
    ushort8 hi = *(const ushort8*)(rowbase + 16 + half * 8);
    ushort16 u = __builtin_shufflevector(lo, hi, 0,1,2,3,4,5,6,7,8,9,10,11,12,13,14,15);
    return __builtin_bit_cast(v16bf, u);
}
static __device__ __forceinline__ v16bf load_b_frag(const u16* rowbase, int half) {
    ushort8 lo = *(const ushort8*)(rowbase + half * 8);
    ushort8 hi = *(const ushort8*)(rowbase + 16 + half * 8);
    ushort16 u = __builtin_shufflevector(lo, hi, 0,1,2,3,4,5,6,7,8,9,10,11,12,13,14,15);
    return __builtin_bit_cast(v16bf, u);
}

// ---------------------------------------------------------------------------
// LDS-staged, double-buffered WMMA GEMM:
//   C[r][n] = alpha * sum_k A[r][k] * Bm[n][k]  (+bias[n]) (+res[r][n])
// A: [rows x K] bf16 row-major; Bm: [ncols x K] bf16 row-major (B^T layout).
// Block = 256 threads = 8 waves (4 row-waves x 2 col-waves), block tile
// 128x128, K-stage 64. Wave tile 32x64 -> 2x4 fragments, 8 WMMA / k-substep.
// A/B stages are copied with GLOBAL_LOAD_ASYNC_TO_LDS_B128 (double buffered).
// LDS rows padded 64 -> 72 u16 for bank-conflict-free ds_load_b128.
// grid = (ncols/128, rows/128); K % 64 == 0.
// ---------------------------------------------------------------------------
#define LDS_PITCH 72

template <bool OUT_BF16, bool TRANS_OUT, bool HAS_BIAS, bool HAS_RES>
__launch_bounds__(256)
__global__ void gemm_bt_wmma(const u16* __restrict__ A, const u16* __restrict__ Bm,
                             void* __restrict__ Out,
                             const float* __restrict__ bias,
                             const float* __restrict__ res,
                             int K, int lda, int ldb, int ldo, float alpha) {
    __shared__ u16 As[2][128 * LDS_PITCH];
    __shared__ u16 Bs[2][128 * LDS_PITCH];

    const int tid  = threadIdx.x;
    const int wave = tid >> 5;
    const int lane = tid & 31;
    const int half = lane >> 4;
    const int l16  = lane & 15;
    const int wm = wave >> 1;       // 0..3 (row wave)
    const int wn = wave & 1;        // 0..1 (col wave)
    const int row0 = blockIdx.y * 128;      // block row origin
    const int col0 = blockIdx.x * 128;      // block col origin
    const int wrow = wm * 32;               // wave row inside block tile
    const int wcol = wn * 64;               // wave col inside block tile

    // Per-thread staging assignment: tile = 128x64 u16 = 1024 chunks of 16B;
    // each thread moves 4 chunks of A and 4 of B per stage (8 async ops).
    //   chunk c (0..1023): row = c>>3, col8 = c&7
    int crow[4], ccol[4];
    unsigned lofsA[4], lofsB[4];
    #pragma unroll
    for (int c = 0; c < 4; ++c) {
        int chunk = tid + c * 256;
        crow[c] = chunk >> 3;
        ccol[c] = (chunk & 7) * 8;
        lofsA[c] = (unsigned)(size_t)(const void*)&As[0][crow[c] * LDS_PITCH + ccol[c]];
        lofsB[c] = (unsigned)(size_t)(const void*)&Bs[0][crow[c] * LDS_PITCH + ccol[c]];
    }
    const unsigned bufstride = (unsigned)(128 * LDS_PITCH * sizeof(u16));

    v8f acc[2][4] = {};

    // Stage loader: async-copy A/B k-panels for stage (k0) into buffer `buf`.
    auto stage = [&](int buf, int k0) {
        #pragma unroll
        for (int c = 0; c < 4; ++c) {
            const u16* ga = A + (size_t)(row0 + crow[c]) * lda + k0 + ccol[c];
            async_copy_b128(lofsA[c] + buf * bufstride, ga);
        }
        #pragma unroll
        for (int c = 0; c < 4; ++c) {
            const u16* gb = Bm + (size_t)(col0 + crow[c]) * ldb + k0 + ccol[c];
            async_copy_b128(lofsB[c] + buf * bufstride, gb);
        }
    };

    stage(0, 0);

    for (int k0 = 0; k0 < K; k0 += 64) {
        const int buf = (k0 >> 6) & 1;
        const bool has_next = (k0 + 64) < K;
        if (has_next) {
            stage(buf ^ 1, k0 + 64);
            wait_async_le8();   // current stage's 8 copies done; next 8 in flight
        } else {
            wait_async_0();
        }
        __syncthreads();

        const u16* as = &As[buf][0];
        const u16* bs = &Bs[buf][0];
        #pragma unroll
        for (int kk = 0; kk < 64; kk += 32) {
            v16bf a0 = load_a_frag(as + (wrow + l16)      * LDS_PITCH + kk, half);
            v16bf a1 = load_a_frag(as + (wrow + 16 + l16) * LDS_PITCH + kk, half);
            v16bf b0 = load_b_frag(bs + (wcol + l16)      * LDS_PITCH + kk, half);
            v16bf b1 = load_b_frag(bs + (wcol + 16 + l16) * LDS_PITCH + kk, half);
            v16bf b2 = load_b_frag(bs + (wcol + 32 + l16) * LDS_PITCH + kk, half);
            v16bf b3 = load_b_frag(bs + (wcol + 48 + l16) * LDS_PITCH + kk, half);
            acc[0][0] = __builtin_amdgcn_wmma_f32_16x16x32_bf16(false, a0, false, b0, (short)0, acc[0][0], false, false);
            acc[0][1] = __builtin_amdgcn_wmma_f32_16x16x32_bf16(false, a0, false, b1, (short)0, acc[0][1], false, false);
            acc[0][2] = __builtin_amdgcn_wmma_f32_16x16x32_bf16(false, a0, false, b2, (short)0, acc[0][2], false, false);
            acc[0][3] = __builtin_amdgcn_wmma_f32_16x16x32_bf16(false, a0, false, b3, (short)0, acc[0][3], false, false);
            acc[1][0] = __builtin_amdgcn_wmma_f32_16x16x32_bf16(false, a1, false, b0, (short)0, acc[1][0], false, false);
            acc[1][1] = __builtin_amdgcn_wmma_f32_16x16x32_bf16(false, a1, false, b1, (short)0, acc[1][1], false, false);
            acc[1][2] = __builtin_amdgcn_wmma_f32_16x16x32_bf16(false, a1, false, b2, (short)0, acc[1][2], false, false);
            acc[1][3] = __builtin_amdgcn_wmma_f32_16x16x32_bf16(false, a1, false, b3, (short)0, acc[1][3], false, false);
        }
        __syncthreads();   // protect buf before it is overwritten next iter
    }

    float bj[4];
    if (HAS_BIAS) {
        #pragma unroll
        for (int j = 0; j < 4; ++j) bj[j] = bias[col0 + wcol + j * 16 + l16];
    }

    #pragma unroll
    for (int i = 0; i < 2; ++i) {
        #pragma unroll
        for (int j = 0; j < 4; ++j) {
            const int colg = col0 + wcol + j * 16 + l16;
            #pragma unroll
            for (int e = 0; e < 8; ++e) {
                const int rowg = row0 + wrow + i * 16 + half * 8 + e;
                float v = acc[i][j][e] * alpha;
                if (HAS_BIAS) v += bj[j];
                if (HAS_RES)  v += res[(size_t)rowg * ldo + colg];
                if (OUT_BF16) {
                    if (TRANS_OUT)
                        ((u16*)Out)[(size_t)colg * ldo + rowg] = f2bf(v);
                    else
                        ((u16*)Out)[(size_t)rowg * ldo + colg] = f2bf(v);
                } else {
                    ((float*)Out)[(size_t)rowg * ldo + colg] = v;
                }
            }
        }
    }
}

// ---------------------------------------------------------------------------
// f32 -> bf16 convert
// ---------------------------------------------------------------------------
__global__ void cvt_f32_bf16(const float* __restrict__ src, u16* __restrict__ dst, int n) {
    int i = blockIdx.x * blockDim.x + threadIdx.x;
    if (i < n) dst[i] = f2bf(src[i]);
}

// f32 [K x N] -> bf16 [N x K] transpose+convert
__global__ void cvt_transpose_f32_bf16(const float* __restrict__ src, u16* __restrict__ dst,
                                       int K, int N) {
    int i = blockIdx.x * blockDim.x + threadIdx.x;
    if (i < K * N) {
        int n = i / K;
        int k = i - n * K;
        dst[(size_t)n * K + k] = f2bf(src[(size_t)k * N + n]);
    }
}

// ---------------------------------------------------------------------------
// Row softmax over bf16 scores, in place. cols == 4096, block = 256 threads,
// 16 elements per thread, one block per row.
// ---------------------------------------------------------------------------
__launch_bounds__(256)
__global__ void softmax_rows_bf16(u16* __restrict__ S, int cols) {
    const int t = threadIdx.x;
    u16* p = S + (size_t)blockIdx.x * cols;
    const int PER = 16;  // 4096 / 256

    float v[PER];
    float mx = -3.4e38f;
    #pragma unroll
    for (int i = 0; i < PER; ++i) {
        v[i] = bf2f(p[t + i * 256]);
        mx = fmaxf(mx, v[i]);
    }

    __shared__ float red[256];
    red[t] = mx; __syncthreads();
    for (int s = 128; s > 0; s >>= 1) {
        if (t < s) red[t] = fmaxf(red[t], red[t + s]);
        __syncthreads();
    }
    mx = red[0]; __syncthreads();

    float sum = 0.f;
    #pragma unroll
    for (int i = 0; i < PER; ++i) {
        v[i] = __expf(v[i] - mx);
        sum += v[i];
    }
    red[t] = sum; __syncthreads();
    for (int s = 128; s > 0; s >>= 1) {
        if (t < s) red[t] += red[t + s];
        __syncthreads();
    }
    const float inv = 1.0f / red[0];

    #pragma unroll
    for (int i = 0; i < PER; ++i)
        p[t + i * 256] = f2bf(v[i] * inv);
}

// ---------------------------------------------------------------------------
// Launch
// ---------------------------------------------------------------------------
extern "C" void kernel_launch(void* const* d_in, const int* in_sizes, int n_in,
                              void* d_out, int out_size, void* d_ws, size_t ws_size,
                              hipStream_t stream) {
    (void)in_sizes; (void)n_in; (void)out_size; (void)ws_size;
    const int Bq = 8192, H = 1024, M = 4096;

    const float* x  = (const float*)d_in[0];
    const float* mb = (const float*)d_in[1];
    const float* Wq = (const float*)d_in[2];
    const float* bq = (const float*)d_in[3];
    const float* Wk = (const float*)d_in[4];
    const float* bk = (const float*)d_in[5];
    const float* Wv = (const float*)d_in[6];
    const float* bv = (const float*)d_in[7];
    float* out = (float*)d_out;

    // workspace layout (bytes)
    char* ws = (char*)d_ws;
    const size_t MB = 1u << 20;
    u16* xb  = (u16*)(ws + 0);          // 8192x1024 bf16 : 16 MB
    u16* mbb = (u16*)(ws + 16 * MB);    // 4096x1024 bf16 :  8 MB
    u16* wqt = (u16*)(ws + 24 * MB);    // 1024x1024 bf16 :  2 MB (transposed)
    u16* wkt = (u16*)(ws + 26 * MB);    //                  2 MB
    u16* wvt = (u16*)(ws + 28 * MB);    //                  2 MB
    u16* Qb  = (u16*)(ws + 30 * MB);    // 8192x1024 bf16 : 16 MB
    u16* Kb  = (u16*)(ws + 46 * MB);    // 4096x1024 bf16 :  8 MB
    u16* Vt  = (u16*)(ws + 54 * MB);    // 1024x4096 bf16 :  8 MB (V transposed)
    u16* Sb  = (u16*)(ws + 62 * MB);    // 8192x4096 bf16 : 64 MB

    // 1) convert inputs to bf16
    {
        int n = Bq * H;
        cvt_f32_bf16<<<(n + 255) / 256, 256, 0, stream>>>(x, xb, n);
        n = M * H;
        cvt_f32_bf16<<<(n + 255) / 256, 256, 0, stream>>>(mb, mbb, n);
        n = H * H;
        cvt_transpose_f32_bf16<<<(n + 255) / 256, 256, 0, stream>>>(Wq, wqt, H, H);
        cvt_transpose_f32_bf16<<<(n + 255) / 256, 256, 0, stream>>>(Wk, wkt, H, H);
        cvt_transpose_f32_bf16<<<(n + 255) / 256, 256, 0, stream>>>(Wv, wvt, H, H);
    }

    // 2) projections
    // Q = x @ Wq + bq        -> bf16 [8192 x 1024]
    gemm_bt_wmma<true, false, true, false>
        <<<dim3(H / 128, Bq / 128), 256, 0, stream>>>(xb, wqt, Qb, bq, nullptr, H, H, H, H, 1.0f);
    // K = mem @ Wk + bk      -> bf16 [4096 x 1024]
    gemm_bt_wmma<true, false, true, false>
        <<<dim3(H / 128, M / 128), 256, 0, stream>>>(mbb, wkt, Kb, bk, nullptr, H, H, H, H, 1.0f);
    // V = mem @ Wv + bv      -> bf16 TRANSPOSED [1024 x 4096] (Vt[h][m])
    gemm_bt_wmma<true, true, true, false>
        <<<dim3(H / 128, M / 128), 256, 0, stream>>>(mbb, wvt, Vt, bv, nullptr, H, H, H, M, 1.0f);

    // 3) scores: S = (Q @ K^T) / 32  -> bf16 [8192 x 4096]
    gemm_bt_wmma<true, false, false, false>
        <<<dim3(M / 128, Bq / 128), 256, 0, stream>>>(Qb, Kb, Sb, nullptr, nullptr, H, H, H, M,
                                                      1.0f / 32.0f);

    // 4) softmax rows in place
    softmax_rows_bf16<<<Bq, 256, 0, stream>>>(Sb, M);

    // 5) out = P @ V + x  -> f32 [8192 x 1024]   (B operand = Vt, so A·B^T again)
    gemm_bt_wmma<false, false, false, true>
        <<<dim3(H / 128, Bq / 128), 256, 0, stream>>>(Sb, Vt, out, nullptr, x, M, M, M, H, 1.0f);
}
